// DecoderStack_47029891891320
// MI455X (gfx1250) — matrix operand order repaired and checked by reading, not compiled, not run on hardware
//
#include <hip/hip_runtime.h>

// ---------------- model dims (fixed by the reference) ----------------
#define D_MODEL 1024
#define N_HEADS 16
#define HEAD_DK 64
#define D_FF    4096
#define VOCAB   32000
#define BATCH   8
#define TDEC    512
#define TP1_LEN 513
#define SENC    1024
#define NLAYERS 6
#define LN_EPS  1e-6f

typedef __attribute__((ext_vector_type(16))) __bf16 v16bf;
typedef __attribute__((ext_vector_type(8)))  float  v8f;

// ---------------- CDNA5 async global->LDS copy (guarded) ----------------
#if defined(__has_builtin)
#if __has_builtin(__builtin_amdgcn_global_load_async_to_lds_b128)
#define HAVE_ASYNC_LDS 1
#endif
#endif

#ifdef HAVE_ASYNC_LDS
// Builtin prototype (from compiler diagnostic): first param is
// `int __attribute__((vector_size(16)))*` in addrspace(1), second in as(3).
typedef int async_v4i __attribute__((vector_size(16)));
typedef __attribute__((address_space(1))) async_v4i* async_gptr;
typedef __attribute__((address_space(3))) async_v4i* async_lptr;
#endif

__device__ __forceinline__ void cp16_g2l(const float* gp, float* lp) {
#ifdef HAVE_ASYNC_LDS
  __builtin_amdgcn_global_load_async_to_lds_b128(
      (async_gptr)(const_cast<float*>(gp)), (async_lptr)lp, 0, 0);
#else
  *reinterpret_cast<float4*>(lp) = *reinterpret_cast<const float4*>(gp);
#endif
}

__device__ __forceinline__ void async_wait_all() {
#ifdef HAVE_ASYNC_LDS
#if __has_builtin(__builtin_amdgcn_s_wait_asynccnt)
  __builtin_amdgcn_s_wait_asynccnt(0);
#else
  asm volatile("s_wait_asynccnt 0" ::: "memory");
#endif
#endif
}

// =====================================================================
// GEMM: C[M,N] = act(A[M,K] @ W[K,N] + bias[N]).  M%64==0, N%64==0, K%32==0.
// fp32 in HBM/LDS, bf16 into v_wmma_f32_16x16x32_bf16, fp32 accumulate.
// Block = 128 threads (4 waves) -> 64x64 output tile (each wave: 16 rows x 4
// 16-col WMMA accumulators).
// =====================================================================
template <bool RELU>
__global__ __launch_bounds__(128) void dec_gemm_wmma(
    const float* __restrict__ A, const float* __restrict__ W,
    const float* __restrict__ bias, float* __restrict__ C,
    int M, int N, int K)
{
  __shared__ __align__(16) float As[64 * 32];
  __shared__ __align__(16) float Bs[32 * 64];

  const int tid  = threadIdx.x;
  const int lane = tid & 31;
  const int wv   = tid >> 5;
  const int lo   = lane & 15;
  const int hi   = lane >> 4;
  const int mbase = blockIdx.y * 64;
  const int nbase = blockIdx.x * 64;

  v8f zero = {};
  v8f acc[4];
#pragma unroll
  for (int j = 0; j < 4; ++j) acc[j] = zero;

  for (int k0 = 0; k0 < K; k0 += 32) {
    // ---- stage A tile 64x32 (async, 512 float4 over 128 threads) ----
#pragma unroll
    for (int i = 0; i < 4; ++i) {
      int idx = tid + i * 128;
      int r = idx >> 3;            // 8 float4 per 32-float row
      int c = (idx & 7) << 2;
      cp16_g2l(A + (size_t)(mbase + r) * K + k0 + c, &As[r * 32 + c]);
    }
    // ---- stage B tile 32x64 ----
#pragma unroll
    for (int i = 0; i < 4; ++i) {
      int idx = tid + i * 128;
      int r = idx >> 4;            // 16 float4 per 64-float row
      int c = (idx & 15) << 2;
      cp16_g2l(W + (size_t)(k0 + r) * N + nbase + c, &Bs[r * 64 + c]);
    }
    async_wait_all();
    __syncthreads();

    // A fragment 16x32 bf16 (contiguous k order, same as B): row = wv*16+lo,
    // K(e) = 16*hi + e
    v16bf afrag;
    const float* arow = &As[(wv * 16 + lo) * 32];
#pragma unroll
    for (int e = 0; e < 16; ++e) {
      int kk = 16 * hi + e;
      afrag[e] = (__bf16)arow[kk];
    }
    // 4 B fragments (N subtiles): B[k][n], k = 16*hi+e, n = j*16+lo
#pragma unroll
    for (int j = 0; j < 4; ++j) {
      v16bf bfrag;
#pragma unroll
      for (int e = 0; e < 16; ++e)
        bfrag[e] = (__bf16)Bs[(hi * 16 + e) * 64 + j * 16 + lo];
      acc[j] = __builtin_amdgcn_wmma_f32_16x16x32_bf16(
          false, afrag, false, bfrag, (short)0, acc[j], false, false);
    }
    __syncthreads();
  }

  // Epilogue: C layout m = e + 8*hi, n = lo
#pragma unroll
  for (int j = 0; j < 4; ++j) {
    const int n = nbase + j * 16 + lo;
    const float bv = bias[n];
#pragma unroll
    for (int e = 0; e < 8; ++e) {
      const int m = mbase + wv * 16 + 8 * hi + e;
      float o = acc[j][e] + bv;
      if (RELU) o = fmaxf(o, 0.0f);
      C[(size_t)m * N + n] = o;
    }
  }
}

// =====================================================================
// Flash attention: one wave per (b, head, 16-query tile).
// Scores and P@V both on WMMA; online softmax in fp32; no score tensor.
// =====================================================================
__global__ __launch_bounds__(32) void dec_flash_attn(
    const float* __restrict__ Q, const float* __restrict__ Kb,
    const float* __restrict__ Vb, float* __restrict__ O,
    int Tq, int Tk, float scale)
{
  __shared__ __align__(16) __bf16 Pt[16 * 32];  // P tile relayout buffer
  const int lane = threadIdx.x;
  const int lo = lane & 15, hi = lane >> 4;
  const int qbase = blockIdx.x * 16;
  const int head  = blockIdx.y;
  const int b     = blockIdx.z;
  const int hb    = head * HEAD_DK;

  // Q fragments (A layout), 2 K-steps of 32 to cover DK=64
  v16bf qf[2];
  {
    const float* qrow = Q + ((size_t)b * Tq + qbase + lo) * D_MODEL + hb;
#pragma unroll
    for (int dstep = 0; dstep < 2; ++dstep)
#pragma unroll
      for (int e = 0; e < 16; ++e) {
        int kk = 16 * hi + e + dstep * 32;
        qf[dstep][e] = (__bf16)qrow[kk];
      }
  }

  v8f zero = {};
  v8f acc[4];
#pragma unroll
  for (int j = 0; j < 4; ++j) acc[j] = zero;
  float m8[8], l8[8];
#pragma unroll
  for (int e = 0; e < 8; ++e) { m8[e] = -3.0e38f; l8[e] = 0.0f; }

  for (int kb0 = 0; kb0 < Tk; kb0 += 32) {
    // ---- S = Q @ K^T for 16x32 score tile (two 16x16 accumulators) ----
    v8f s0 = zero, s1 = zero;
#pragma unroll
    for (int dstep = 0; dstep < 2; ++dstep) {
      v16bf kf0, kf1;
      const float* kr0 = Kb + ((size_t)b * Tk + kb0 + lo) * D_MODEL
                         + hb + dstep * 32 + hi * 16;
      const float* kr1 = Kb + ((size_t)b * Tk + kb0 + 16 + lo) * D_MODEL
                         + hb + dstep * 32 + hi * 16;
#pragma unroll
      for (int e = 0; e < 16; ++e) {
        kf0[e] = (__bf16)kr0[e];
        kf1[e] = (__bf16)kr1[e];
      }
      s0 = __builtin_amdgcn_wmma_f32_16x16x32_bf16(
          false, qf[dstep], false, kf0, (short)0, s0, false, false);
      s1 = __builtin_amdgcn_wmma_f32_16x16x32_bf16(
          false, qf[dstep], false, kf1, (short)0, s1, false, false);
    }

    // ---- online softmax (row stats live per element e = row e+8*hi) ----
    float alpha[8];
#pragma unroll
    for (int e = 0; e < 8; ++e) {
      float a0 = s0[e] * scale, a1 = s1[e] * scale;
      float mx = fmaxf(a0, a1);
      mx = fmaxf(mx, __shfl_xor(mx, 1, 32));
      mx = fmaxf(mx, __shfl_xor(mx, 2, 32));
      mx = fmaxf(mx, __shfl_xor(mx, 4, 32));
      mx = fmaxf(mx, __shfl_xor(mx, 8, 32));
      float mnew = fmaxf(m8[e], mx);
      float p0 = __expf(a0 - mnew), p1 = __expf(a1 - mnew);
      float r = p0 + p1;
      r += __shfl_xor(r, 1, 32);
      r += __shfl_xor(r, 2, 32);
      r += __shfl_xor(r, 4, 32);
      r += __shfl_xor(r, 8, 32);
      float al = __expf(m8[e] - mnew);
      l8[e] = l8[e] * al + r;
      m8[e] = mnew;
      alpha[e] = al;
      s0[e] = p0; s1[e] = p1;
    }
#pragma unroll
    for (int j = 0; j < 4; ++j)
#pragma unroll
      for (int e = 0; e < 8; ++e) acc[j][e] *= alpha[e];

    // ---- relayout P (C layout -> A layout) via LDS ----
    __syncthreads();
#pragma unroll
    for (int e = 0; e < 8; ++e) {
      Pt[(8 * hi + e) * 32 + lo]      = (__bf16)s0[e];
      Pt[(8 * hi + e) * 32 + 16 + lo] = (__bf16)s1[e];
    }
    __syncthreads();
    v16bf pf;
#pragma unroll
    for (int e = 0; e < 16; ++e) {
      int kk = 16 * hi + e;
      pf[e] = Pt[lo * 32 + kk];
    }

    // ---- acc += P @ V (4 DK subtiles of 16) ----
#pragma unroll
    for (int j = 0; j < 4; ++j) {
      v16bf vf;
      const float* vcol = Vb + ((size_t)b * Tk + kb0 + hi * 16) * D_MODEL
                          + hb + j * 16 + lo;
#pragma unroll
      for (int e = 0; e < 16; ++e)
        vf[e] = (__bf16)vcol[(size_t)e * D_MODEL];
      acc[j] = __builtin_amdgcn_wmma_f32_16x16x32_bf16(
          false, pf, false, vf, (short)0, acc[j], false, false);
    }
  }

  // ---- write O = acc / rowsum ----
#pragma unroll
  for (int j = 0; j < 4; ++j)
#pragma unroll
    for (int e = 0; e < 8; ++e)
      O[((size_t)b * Tq + qbase + 8 * hi + e) * D_MODEL + hb + j * 16 + lo]
          = acc[j][e] / l8[e];
}

// =====================================================================
// Y = gamma * (X+R - mean) / (sqrt(var_unbiased) + eps) + beta, per row of D
// =====================================================================
__global__ __launch_bounds__(256) void dec_add_norm(
    const float* __restrict__ X, const float* __restrict__ R,
    const float* __restrict__ ga, const float* __restrict__ gb,
    float* __restrict__ Y)
{
  __shared__ float red[8];
  const int row = blockIdx.x;
  const int tid = threadIdx.x;
  const size_t base = (size_t)row * D_MODEL;
  float v[4];
  float s = 0.0f;
#pragma unroll
  for (int i = 0; i < 4; ++i) {
    int d = tid + i * 256;
    v[i] = X[base + d] + R[base + d];
    s += v[i];
  }
#pragma unroll
  for (int m = 16; m >= 1; m >>= 1) s += __shfl_xor(s, m, 32);
  if ((tid & 31) == 0) red[tid >> 5] = s;
  __syncthreads();
  if (tid == 0) { float t = 0; for (int i = 0; i < 8; ++i) t += red[i]; red[0] = t; }
  __syncthreads();
  const float mean = red[0] * (1.0f / D_MODEL);
  __syncthreads();
  float vs = 0.0f;
#pragma unroll
  for (int i = 0; i < 4; ++i) { float d0 = v[i] - mean; vs += d0 * d0; }
#pragma unroll
  for (int m = 16; m >= 1; m >>= 1) vs += __shfl_xor(vs, m, 32);
  if ((tid & 31) == 0) red[tid >> 5] = vs;
  __syncthreads();
  if (tid == 0) { float t = 0; for (int i = 0; i < 8; ++i) t += red[i]; red[0] = t; }
  __syncthreads();
  const float var = red[0] * (1.0f / (D_MODEL - 1));   // torch unbiased std
  const float inv = 1.0f / (sqrtf(var) + LN_EPS);
#pragma unroll
  for (int i = 0; i < 4; ++i) {
    int d = tid + i * 256;
    Y[base + d] = ga[d] * (v[i] - mean) * inv + gb[d];
  }
}

// =====================================================================
__global__ __launch_bounds__(256) void dec_embed(
    const int* __restrict__ x, const float* __restrict__ emb,
    const float* __restrict__ pe, float* __restrict__ h)
{
  const int bt = blockIdx.x;
  const int b = bt >> 9, t = bt & 511;
  const int tok = x[b * TP1_LEN + t];
  const size_t hb = (size_t)bt * D_MODEL;
  const size_t eb = (size_t)tok * D_MODEL;
  const size_t pb = (size_t)t * D_MODEL;
#pragma unroll
  for (int i = 0; i < 4; ++i) {
    int d = threadIdx.x + i * 256;
    h[hb + d] = emb[eb + d] + pe[pb + d];
  }
}

// =====================================================================
__global__ __launch_bounds__(256) void dec_log_softmax(float* __restrict__ P)
{
  __shared__ float red[8];
  const int tid = threadIdx.x;
  float* p = P + (size_t)blockIdx.x * VOCAB;
  float mx = -3.0e38f;
  for (int i = tid; i < VOCAB; i += 256) mx = fmaxf(mx, p[i]);
#pragma unroll
  for (int m = 16; m >= 1; m >>= 1) mx = fmaxf(mx, __shfl_xor(mx, m, 32));
  if ((tid & 31) == 0) red[tid >> 5] = mx;
  __syncthreads();
  if (tid == 0) { float t = red[0]; for (int i = 1; i < 8; ++i) t = fmaxf(t, red[i]); red[0] = t; }
  __syncthreads();
  const float M = red[0];
  __syncthreads();
  float s = 0.0f;
  for (int i = tid; i < VOCAB; i += 256) s += __expf(p[i] - M);
#pragma unroll
  for (int m = 16; m >= 1; m >>= 1) s += __shfl_xor(s, m, 32);
  if ((tid & 31) == 0) red[tid >> 5] = s;
  __syncthreads();
  if (tid == 0) { float t = 0; for (int i = 0; i < 8; ++i) t += red[i]; red[0] = t; }
  __syncthreads();
  const float lse = M + logf(red[0]);
  for (int i = tid; i < VOCAB; i += 256) p[i] -= lse;
}

// =====================================================================
extern "C" void kernel_launch(void* const* d_in, const int* in_sizes, int n_in,
                              void* d_out, int out_size, void* d_ws, size_t ws_size,
                              hipStream_t stream)
{
  const int*   x    = (const int*)  d_in[0];
  const float* enc  = (const float*)d_in[1];
  const float* emb  = (const float*)d_in[2];
  const float* pe   = (const float*)d_in[3];
  const float* a1w  = (const float*)d_in[4];
  const float* a1b  = (const float*)d_in[5];
  const float* a2w  = (const float*)d_in[6];
  const float* a2b  = (const float*)d_in[7];
  const float* na   = (const float*)d_in[8];
  const float* nb   = (const float*)d_in[9];
  const float* f1w  = (const float*)d_in[10];
  const float* f1b  = (const float*)d_in[11];
  const float* f2w  = (const float*)d_in[12];
  const float* f2b  = (const float*)d_in[13];
  const float* fcw  = (const float*)d_in[14];
  const float* fcb  = (const float*)d_in[15];
  float* out = (float*)d_out;

  const size_t M_dec = (size_t)BATCH * TDEC;   // 4096 rows
  const size_t M_enc = (size_t)BATCH * SENC;   // 8192 rows
  float* h  = (float*)d_ws;                    // [M_dec, D]
  float* qb = h  + M_dec * D_MODEL;            // [M_dec, D]
  float* kb = qb + M_dec * D_MODEL;            // [M_enc, D]
  float* vb = kb + M_enc * D_MODEL;            // [M_enc, D]
  float* ob = vb + M_enc * D_MODEL;            // [M_dec, D]
  float* ff = ob + M_dec * D_MODEL;            // [M_dec, DFF]

  auto gemm = [&](const float* A, const float* W, const float* B, float* C,
                  int M, int N, int K, bool relu) {
    dim3 grid((unsigned)(N / 64), (unsigned)(M / 64)), block(128);
    if (relu) dec_gemm_wmma<true ><<<grid, block, 0, stream>>>(A, W, B, C, M, N, K);
    else      dec_gemm_wmma<false><<<grid, block, 0, stream>>>(A, W, B, C, M, N, K);
  };
  auto addnorm = [&](const float* X, const float* R, int l, int which) {
    dec_add_norm<<<dim3((unsigned)M_dec), dim3(256), 0, stream>>>(
        X, R, na + ((size_t)l * 3 + which) * D_MODEL,
        nb + ((size_t)l * 3 + which) * D_MODEL, h);
  };

  dec_embed<<<dim3((unsigned)M_dec), dim3(256), 0, stream>>>(x, emb, pe, h);

  const size_t DD = (size_t)D_MODEL * D_MODEL;
  const float kscale = 0.125f;  // 1/sqrt(DK=64)
  for (int l = 0; l < NLAYERS; ++l) {
    // ---------------- self attention ----------------
    const float* w  = a1w + (size_t)l * 4 * DD;
    const float* bb = a1b + (size_t)l * 4 * D_MODEL;
    gemm(h, w + 0 * DD, bb + 0 * D_MODEL, qb, (int)M_dec, D_MODEL, D_MODEL, false);
    gemm(h, w + 1 * DD, bb + 1 * D_MODEL, kb, (int)M_dec, D_MODEL, D_MODEL, false);
    gemm(h, w + 2 * DD, bb + 2 * D_MODEL, vb, (int)M_dec, D_MODEL, D_MODEL, false);
    dec_flash_attn<<<dim3(TDEC / 16, N_HEADS, BATCH), dim3(32), 0, stream>>>(
        qb, kb, vb, ob, TDEC, TDEC, kscale);
    gemm(ob, w + 3 * DD, bb + 3 * D_MODEL, qb, (int)M_dec, D_MODEL, D_MODEL, false);
    addnorm(qb, h, l, 0);
    // ---------------- cross attention ----------------
    const float* w2  = a2w + (size_t)l * 4 * DD;
    const float* bb2 = a2b + (size_t)l * 4 * D_MODEL;
    gemm(h,   w2 + 0 * DD, bb2 + 0 * D_MODEL, qb, (int)M_dec, D_MODEL, D_MODEL, false);
    gemm(enc, w2 + 1 * DD, bb2 + 1 * D_MODEL, kb, (int)M_enc, D_MODEL, D_MODEL, false);
    gemm(enc, w2 + 2 * DD, bb2 + 2 * D_MODEL, vb, (int)M_enc, D_MODEL, D_MODEL, false);
    dec_flash_attn<<<dim3(TDEC / 16, N_HEADS, BATCH), dim3(32), 0, stream>>>(
        qb, kb, vb, ob, TDEC, SENC, kscale);
    gemm(ob, w2 + 3 * DD, bb2 + 3 * D_MODEL, qb, (int)M_dec, D_MODEL, D_MODEL, false);
    addnorm(qb, h, l, 1);
    // ---------------- feed forward ----------------
    gemm(h,  f1w + (size_t)l * D_MODEL * D_FF, f1b + (size_t)l * D_FF, ff,
         (int)M_dec, D_FF, D_MODEL, true);
    gemm(ff, f2w + (size_t)l * D_FF * D_MODEL, f2b + (size_t)l * D_MODEL, ob,
         (int)M_dec, D_MODEL, D_FF, false);
    addnorm(ob, h, l, 2);
  }

  // ---------------- final projection + log-softmax ----------------
  gemm(h, fcw, fcb, out, (int)M_dec, VOCAB, D_MODEL, false);
  dec_log_softmax<<<dim3((unsigned)M_dec), dim3(256), 0, stream>>>(out);

  (void)in_sizes; (void)n_in; (void)out_size; (void)ws_size;
}
